// MyGCN_58179626992409
// MI455X (gfx1250) — compile-verified, matrix operand order checked by reading, not run on hardware
//
#include <hip/hip_runtime.h>
#include <hip/hip_bf16.h>

// ---------------------------------------------------------------------------
// GCN (3x GCNConv) for MI455X / gfx1250, wave32.
//   * Dense GEMMs: V_WMMA_F32_16X16X32_BF16, one wave per 16x64 output strip
//     (4 column tiles share each A fragment -> 4x less A traffic, 4
//     independent WMMAs issue back-to-back per K-chunk).
//   * Edge aggregation: float4 gather + global f32 atomics (memory-bound,
//     ~3.3 GB total -> ~140us at 23.3 TB/s; GEMM operands are L2-resident).
// ---------------------------------------------------------------------------

typedef __attribute__((ext_vector_type(16))) __bf16 v16bf;
typedef __attribute__((ext_vector_type(8)))  float  v8f;

union FragU {
    uint4 q[2];
    v16bf v;
};

// ---------------- simple utility kernels ----------------

__global__ void k_zero_f32(float* __restrict__ p, long long n) {
    long long i = (long long)blockIdx.x * blockDim.x + threadIdx.x;
    if (i < n) p[i] = 0.0f;
}

__global__ void k_degree(const int* __restrict__ col, float* __restrict__ deg, int E) {
    int e = blockIdx.x * blockDim.x + threadIdx.x;
    if (e < E) atomicAdd(&deg[col[e]], 1.0f);
}

__global__ void k_rsqrt(float* __restrict__ d, int N) {
    int i = blockIdx.x * blockDim.x + threadIdx.x;
    if (i < N) d[i] = rsqrtf(d[i] + 1.0f);
}

__global__ void k_f32_to_bf16(const float* __restrict__ in, __bf16* __restrict__ out, long long n) {
    long long i = (long long)blockIdx.x * blockDim.x + threadIdx.x;
    if (i < n) out[i] = (__bf16)in[i];
}

// W[K x F] (row-major, as in x @ W) -> Wt[F x K] bf16 so that column n of W is
// a contiguous row of Wt (contiguous 16B fragment loads for the B operand).
__global__ void k_transpose_w(const float* __restrict__ W, __bf16* __restrict__ Wt, int K, int F) {
    int idx = blockIdx.x * blockDim.x + threadIdx.x;
    if (idx < K * F) {
        int fi = idx / F;   // 0..K-1
        int fo = idx % F;   // 0..F-1
        Wt[(size_t)fo * K + fi] = (__bf16)W[idx];
    }
}

// ---------------- WMMA GEMM: D[M x F] = A[M x K](bf16) * Wt[F x K](bf16)^T ----
// One wave per 16x64 output strip (4 column tiles). K multiple of 32, F
// multiple of 64, M multiple of 16 (N=100000 -> 6250 row tiles exactly).

__global__ void k_gemm_bf16_wmma(const __bf16* __restrict__ A,
                                 const __bf16* __restrict__ Bt,
                                 float* __restrict__ D,
                                 int mTiles, int K, int F) {
    const int wavesPerBlock = blockDim.x >> 5;
    const int waveId = blockIdx.x * wavesPerBlock + (threadIdx.x >> 5);
    const int nGroups = F >> 6;                    // groups of 4 n-tiles
    if (waveId >= mTiles * nGroups) return;        // wave-uniform: EXEC all-ones

    const int tm   = waveId / nGroups;
    const int tg   = waveId % nGroups;
    const int lane = threadIdx.x & 31;
    const int half = lane >> 4;      // K sub-range selector
    const int l15  = lane & 15;      // A row / B column within tile

    const size_t arow = (size_t)(tm * 16 + l15) * (size_t)K;
    const size_t brow = (size_t)(tg * 64 + l15) * (size_t)K;   // +j*16*K per tile
    const size_t bstep = (size_t)16 * (size_t)K;

    v8f acc[4];
#pragma unroll
    for (int j = 0; j < 4; ++j) acc[j] = (v8f){0.f, 0.f, 0.f, 0.f, 0.f, 0.f, 0.f, 0.f};

    for (int k0 = 0; k0 < K; k0 += 32) {
        // per-lane A fragment (ISA 16-bit A 16x32 layout): elems 0..7 ->
        // K=k0+8*half+i, elems 8..15 -> K=k0+16+8*half+i; two 16B loads.
        FragU a;
        a.q[0] = *reinterpret_cast<const uint4*>(A + arow + k0 + 8 * half);
        a.q[1] = *reinterpret_cast<const uint4*>(A + arow + k0 + 16 + 8 * half);
        FragU b[4];
#pragma unroll
        for (int j = 0; j < 4; ++j) {
            const __bf16* bp = Bt + brow + (size_t)j * bstep + k0;
            b[j].q[0] = *reinterpret_cast<const uint4*>(bp + 8 * half);
            b[j].q[1] = *reinterpret_cast<const uint4*>(bp + 16 + 8 * half);
        }
#pragma unroll
        for (int j = 0; j < 4; ++j) {
            acc[j] = __builtin_amdgcn_wmma_f32_16x16x32_bf16(
                /*neg_a=*/false, a.v, /*neg_b=*/false, b[j].v,
                /*c_mod=*/(short)0, acc[j], /*reuse_a=*/false, /*reuse_b=*/false);
        }
    }

    // C/D layout: VGPR v -> row (v + 8*half), col = lane&15
#pragma unroll
    for (int j = 0; j < 4; ++j) {
        float* drow = D + (size_t)(tm * 16 + 8 * half) * (size_t)F
                        + (size_t)(tg * 64 + j * 16 + l15);
#pragma unroll
        for (int v = 0; v < 8; ++v) drow[(size_t)v * F] = acc[j][v];
    }
}

// ---------------- edge gather/scale/scatter-add ----------------
// One wave per edge; lanes stride float4 chunks of the feature dimension.

__global__ void k_edge_scatter(const float* __restrict__ h,
                               const int* __restrict__ row,
                               const int* __restrict__ col,
                               const float* __restrict__ dis,
                               float* __restrict__ agg,
                               int E, int F) {
    int wave = (int)(((long long)blockIdx.x * blockDim.x + threadIdx.x) >> 5);
    int lane = threadIdx.x & 31;
    if (wave >= E) return;
    int r = row[wave];          // lane-uniform address -> single request
    int c = col[wave];
    float nrm = dis[r] * dis[c];
    const float4* hs = reinterpret_cast<const float4*>(h + (size_t)r * F);
    float* ad = agg + (size_t)c * F;
    const int F4 = F >> 2;
    for (int f4 = lane; f4 < F4; f4 += 32) {
        float4 v = hs[f4];                 // global_load_b128 gather
        int f = f4 << 2;
        atomicAdd(ad + f + 0, v.x * nrm);
        atomicAdd(ad + f + 1, v.y * nrm);
        atomicAdd(ad + f + 2, v.z * nrm);
        atomicAdd(ad + f + 3, v.w * nrm);
    }
}

// ---------------- combine: out = agg + h*dis^2 + b  (+ReLU, +bf16 re-encode) --

__global__ void k_combine(const float* __restrict__ agg,
                          const float* __restrict__ h,
                          const float* __restrict__ dis,
                          const float* __restrict__ bias,
                          float* __restrict__ out_f32,
                          __bf16* __restrict__ out_bf16,
                          int N, int F, int do_relu) {
    long long idx = (long long)blockIdx.x * blockDim.x + threadIdx.x;
    if (idx >= (long long)N * F) return;
    int i = (int)(idx / F);
    int f = (int)(idx % F);
    float d = dis[i];
    float v = agg[idx] + h[idx] * d * d + bias[f];
    if (do_relu) v = fmaxf(v, 0.0f);
    if (out_f32)  out_f32[idx] = v;
    if (out_bf16) out_bf16[idx] = (__bf16)v;
}

// ---------------------------------------------------------------------------

extern "C" void kernel_launch(void* const* d_in, const int* in_sizes, int n_in,
                              void* d_out, int out_size, void* d_ws, size_t ws_size,
                              hipStream_t stream) {
    const int IN_DIM = 64, HID_DIM = 128, OUT_DIM = 64;
    const int N = in_sizes[0] / IN_DIM;
    const int E = in_sizes[1] / 2;

    const float* x  = (const float*)d_in[0];
    const int*   ei = (const int*)d_in[1];
    const int*   row = ei;          // edge_index[0]
    const int*   col = ei + E;      // edge_index[1]
    const float* W1 = (const float*)d_in[2];
    const float* b1 = (const float*)d_in[3];
    const float* W2 = (const float*)d_in[4];
    const float* b2 = (const float*)d_in[5];
    const float* W3 = (const float*)d_in[6];
    const float* b3 = (const float*)d_in[7];
    float* out = (float*)d_out;

    // workspace carve-up (256B aligned chunks)
    char* ws = (char*)d_ws;
    size_t off = 0;
    auto carve = [&](size_t bytes) {
        void* p = ws + off;
        off += (bytes + 255) & ~(size_t)255;
        return p;
    };
    float*  dis = (float*)carve((size_t)N * 4);                     // deg -> rsqrt
    float*  h   = (float*)carve((size_t)N * HID_DIM * 4);           // GEMM output
    float*  agg = (float*)carve((size_t)N * HID_DIM * 4);           // scatter target
    __bf16* act = (__bf16*)carve((size_t)N * HID_DIM * 2);          // bf16 activations
    __bf16* Wt  = (__bf16*)carve((size_t)HID_DIM * HID_DIM * 2);    // bf16 W^T
    (void)ws_size; (void)n_in; (void)out_size;

    const int T = 256;
    auto blocks = [](long long n, int t) { return (unsigned)((n + t - 1) / t); };

    // ---- degree / normalization ----
    k_zero_f32<<<blocks(N, T), T, 0, stream>>>(dis, N);
    k_degree<<<blocks(E, T), T, 0, stream>>>(col, dis, E);
    k_rsqrt<<<blocks(N, T), T, 0, stream>>>(dis, N);

    // ---- x -> bf16 activations ----
    k_f32_to_bf16<<<blocks((long long)N * IN_DIM, T), T, 0, stream>>>(x, act, (long long)N * IN_DIM);

    const int mTiles = N / 16;  // N = 100000 -> exactly 6250 row tiles

    struct Layer { const float* W; const float* b; int K; int F; int relu; };
    Layer layers[3] = {
        { W1, b1, IN_DIM,  HID_DIM, 1 },
        { W2, b2, HID_DIM, HID_DIM, 1 },
        { W3, b3, HID_DIM, OUT_DIM, 0 },
    };

    for (int L = 0; L < 3; ++L) {
        const Layer& ly = layers[L];
        // W[K x F] -> Wt[F x K] bf16
        k_transpose_w<<<blocks(ly.K * ly.F, T), T, 0, stream>>>(ly.W, Wt, ly.K, ly.F);

        // h = act @ W via WMMA: one wave per 16x64 strip, 8 waves/block
        long long strips = (long long)mTiles * (ly.F >> 6);
        k_gemm_bf16_wmma<<<blocks(strips * 32, T), T, 0, stream>>>(act, Wt, h, mTiles, ly.K, ly.F);

        // agg = 0 ; scatter-add normalized messages
        long long nf = (long long)N * ly.F;
        k_zero_f32<<<blocks(nf, T), T, 0, stream>>>(agg, nf);
        k_edge_scatter<<<blocks((long long)E * 32, T), T, 0, stream>>>(h, row, col, dis, agg, E, ly.F);

        // combine (+bias, self-loop, ReLU) -> next bf16 activations or final f32
        if (L < 2) {
            k_combine<<<blocks(nf, T), T, 0, stream>>>(agg, h, dis, ly.b,
                                                       (float*)nullptr, act, N, ly.F, ly.relu);
        } else {
            k_combine<<<blocks(nf, T), T, 0, stream>>>(agg, h, dis, ly.b,
                                                       out, (__bf16*)nullptr, N, ly.F, ly.relu);
        }
    }
}
